// MultiHeadSelfAttention_37194416783949
// MI455X (gfx1250) — matrix operand order repaired and checked by reading, not compiled, not run on hardware
//
#include <hip/hip_runtime.h>
#include <hip/hip_bf16.h>

typedef __attribute__((ext_vector_type(16))) _Float16 v16h;
typedef __attribute__((ext_vector_type(8)))  _Float16 v8h;
typedef __attribute__((ext_vector_type(4)))  _Float16 v4h;
typedef __attribute__((ext_vector_type(8)))  float    v8f;
typedef __attribute__((ext_vector_type(4)))  unsigned int v4u;
typedef __attribute__((ext_vector_type(8)))  int      v8i;
typedef __attribute__((ext_vector_type(4)))  int      v4i;

#define B_    4
#define S_    2048
#define D_    512
#define H_    8
#define DK_   64
#define M_    (B_ * S_)      // 8192
#define LDSPAD 40            // 32 f16 data + 8 pad -> 80B stride (TDM pad 3/3)
#define KPAD   72            // 64 f16 data + 8 pad -> 144B stride (TDM pad 4/3)

#if defined(__has_builtin)
#  if __has_builtin(__builtin_amdgcn_tensor_load_to_lds)
#    define HAS_TDM 1
#  endif
#endif
#ifndef HAS_TDM
#  define HAS_TDM 0
#endif

static __device__ __forceinline__ v8f wmma16(v16h a, v16h b, v8f c) {
    return __builtin_amdgcn_wmma_f32_16x16x32_f16(false, a, false, b, (short)0, c, false, false);
}
static __device__ __forceinline__ v16h mk16(v8h lo, v8h hi) {
    return __builtin_shufflevector(lo, hi, 0,1,2,3,4,5,6,7,8,9,10,11,12,13,14,15);
}
static __device__ __forceinline__ v8f vzero8f() {
    v8f z;
#pragma unroll
    for (int i = 0; i < 8; ++i) z[i] = 0.0f;
    return z;
}

static __device__ __forceinline__ unsigned ldsoff(const void* p) {
    // LDS flat addresses carry the byte offset in the low 32 bits
    return (unsigned)(size_t)p;
}

#if HAS_TDM
// 2-D f16 tile load via Tensor Data Mover.  pad_int/pad_amt are the D# codes:
// pad_int c => pad after 2^(c+1) DWORDs; pad_amt c => insert (c+1) DWORDs.
static __device__ __forceinline__ void tdm_load_2d(
    unsigned lds_byte, const _Float16* g, unsigned tile_w, unsigned tile_h,
    unsigned stride_elems, unsigned pad_int, unsigned pad_amt)
{
    unsigned long long ga = (unsigned long long)(size_t)g;
    v4u g0;
    g0[0] = 1u;                                   // count=1, user mode
    g0[1] = lds_byte;                             // lds_addr
    g0[2] = (unsigned)ga;                         // global_addr[31:0]
    g0[3] = (unsigned)((ga >> 32) & 0x1FFFFFFu) | (2u << 30);  // addr[56:32]|type=2
    v8i g1;
    g1[0] = (int)((1u << 16) | (1u << 20) | (pad_int << 22) | (pad_amt << 25));
    g1[1] = (int)((tile_w & 0xFFFFu) << 16);      // tensor_dim0[15:0]
    g1[2] = (int)(((tile_w >> 16) & 0xFFFFu) | ((tile_h & 0xFFFFu) << 16));
    g1[3] = (int)(((tile_h >> 16) & 0xFFFFu) | ((tile_w & 0xFFFFu) << 16)); // tile_dim0
    g1[4] = (int)(tile_h & 0xFFFFu);              // tile_dim1 (tile_dim2=0)
    g1[5] = (int)stride_elems;                    // tensor_dim0_stride[31:0]
    g1[6] = 0;
    g1[7] = 0;
    v4i z4; z4[0] = z4[1] = z4[2] = z4[3] = 0;
#if __clang_major__ >= 23
    v8i z8;
#pragma unroll
    for (int i = 0; i < 8; ++i) z8[i] = 0;
    __builtin_amdgcn_tensor_load_to_lds(g0, g1, z4, z4, z8, 0);
#else
    __builtin_amdgcn_tensor_load_to_lds(g0, g1, z4, z4, 0);
#endif
}
#endif

// Wave-scalar wait; free (TENSORcnt==0) for waves that issued nothing, so it
// is executed unconditionally to avoid EXEC save/restore branches.
static __device__ __forceinline__ void tdm_wait0() {
#if defined(__has_builtin) && __has_builtin(__builtin_amdgcn_s_wait_tensorcnt)
    __builtin_amdgcn_s_wait_tensorcnt(0);
#else
    asm volatile("s_wait_tensorcnt 0x0" ::: "memory");
#endif
}

// ---------------------------------------------------------------------------
// Kernel 0a/0b: fp32 -> f16 conversion passes (x and the four weights) so the
// GEMM kernels can stage raw f16 tiles with the TDM.
// ---------------------------------------------------------------------------
__global__ __launch_bounds__(256) void cvt_x_f16(const float* __restrict__ x,
                                                 _Float16* __restrict__ Xh) {
    const size_t i = ((size_t)blockIdx.x * 256 + threadIdx.x) * 4;
    float4 f = *(const float4*)(x + i);
    v4h h;
    h[0] = (_Float16)f.x; h[1] = (_Float16)f.y;
    h[2] = (_Float16)f.z; h[3] = (_Float16)f.w;
    *(v4h*)(Xh + i) = h;
}

__global__ __launch_bounds__(256) void cvt_w_f16(
    const float* __restrict__ Wq, const float* __restrict__ Wk,
    const float* __restrict__ Wv, const float* __restrict__ Wo,
    _Float16* __restrict__ Wqh, _Float16* __restrict__ Wkh,
    _Float16* __restrict__ Wvh, _Float16* __restrict__ Woh) {
    const int m = blockIdx.y;
    const float* src = (m == 0) ? Wq : (m == 1) ? Wk : (m == 2) ? Wv : Wo;
    _Float16* dst = (m == 0) ? Wqh : (m == 1) ? Wkh : (m == 2) ? Wvh : Woh;
    const size_t i = ((size_t)blockIdx.x * 256 + threadIdx.x) * 4;
    float4 f = *(const float4*)(src + i);
    v4h h;
    h[0] = (_Float16)f.x; h[1] = (_Float16)f.y;
    h[2] = (_Float16)f.z; h[3] = (_Float16)f.w;
    *(v4h*)(dst + i) = h;
}

// ---------------------------------------------------------------------------
// Kernel 1: fused QKV projection.  y = x @ W^T + b.
// grid = (M/64, 1536/128), block = 256 (8 waves).  Q,K -> (B,H,S,dk) f16,
// V -> transposed (B,H,dk,S).  Tiles staged by TDM, double buffered.
// ---------------------------------------------------------------------------
__global__ __launch_bounds__(256) void qkv_proj_wmma(
    const _Float16* __restrict__ Xh,
    const _Float16* __restrict__ Wqh, const _Float16* __restrict__ Wkh,
    const _Float16* __restrict__ Wvh,
    const float* __restrict__ bq, const float* __restrict__ bk,
    const float* __restrict__ bv,
    _Float16* __restrict__ Qh, _Float16* __restrict__ Kh,
    _Float16* __restrict__ VTh)
{
    const int tid    = threadIdx.x;
    const int lane   = tid & 31;
    const int wid    = tid >> 5;
    const int lane16 = lane & 15;
    const int half   = lane >> 4;

    const int m0  = blockIdx.x * 64;
    const int gn0 = blockIdx.y * 128;
    const int mat = gn0 >> 9;              // 0=Q, 1=K, 2=V
    const int nl0 = gn0 & 511;

    const _Float16* W = (mat == 0) ? Wqh : (mat == 1) ? Wkh : Wvh;
    const float* bias = (mat == 0) ? bq : (mat == 1) ? bk : bv;

    __shared__ __align__(16) _Float16 As[2][64 * LDSPAD];
    __shared__ __align__(16) _Float16 Bs[2][128 * LDSPAD];

    const int wm = wid >> 2, wn = wid & 3;
    const int rm = wm * 32,  cn = wn * 32;

    v8f acc[2][2];
#pragma unroll
    for (int i = 0; i < 2; ++i)
#pragma unroll
        for (int j = 0; j < 2; ++j) acc[i][j] = vzero8f();

#if HAS_TDM
    if (wid == 0) {
        tdm_load_2d(ldsoff(As[0]), Xh + (size_t)m0 * 512,  32,  64, 512, 3, 3);
        tdm_load_2d(ldsoff(Bs[0]), W  + (size_t)nl0 * 512, 32, 128, 512, 3, 3);
    }
    tdm_wait0();
    __syncthreads();
#endif
    for (int it = 0; it < 16; ++it) {
        const int kk = it * 32;
#if HAS_TDM
        const int cur = it & 1;
        if (it + 1 < 16 && wid == 0) {   // prefetch next tile into other buffer
            tdm_load_2d(ldsoff(As[cur ^ 1]), Xh + (size_t)m0 * 512 + kk + 32,
                        32, 64, 512, 3, 3);
            tdm_load_2d(ldsoff(Bs[cur ^ 1]), W + (size_t)nl0 * 512 + kk + 32,
                        32, 128, 512, 3, 3);
        }
        const _Float16* Asb = As[cur];
        const _Float16* Bsb = Bs[cur];
#else
        __syncthreads();
        {   // A tile: 64 rows x 32 f16, 4 thr/row
            const int row = tid >> 2, kj = (tid & 3) * 8;
            *(v8h*)(As[0] + row * LDSPAD + kj) =
                *(const v8h*)(Xh + (size_t)(m0 + row) * 512 + kk + kj);
        }
        {   // B tile: 128 rows x 32 f16, 2 thr/row
            const int row = tid >> 1, kj = (tid & 1) * 16;
            const _Float16* src = W + (size_t)(nl0 + row) * 512 + kk + kj;
            *(v8h*)(Bs[0] + row * LDSPAD + kj)     = *(const v8h*)src;
            *(v8h*)(Bs[0] + row * LDSPAD + kj + 8) = *(const v8h*)(src + 8);
        }
        __syncthreads();
        const _Float16* Asb = As[0];
        const _Float16* Bsb = Bs[0];
#endif
        v16h a[2], b[2];
#pragma unroll
        for (int tm = 0; tm < 2; ++tm) {
            const _Float16* p = Asb + (rm + tm * 16 + lane16) * LDSPAD;
            a[tm] = mk16(*(const v8h*)(p + half * 8),
                         *(const v8h*)(p + 16 + half * 8));
        }
#pragma unroll
        for (int tn = 0; tn < 2; ++tn) {
            const _Float16* p = Bsb + (cn + tn * 16 + lane16) * LDSPAD + half * 8;
            b[tn] = mk16(*(const v8h*)p, *(const v8h*)(p + 16));
        }
#pragma unroll
        for (int tm = 0; tm < 2; ++tm)
#pragma unroll
            for (int tn = 0; tn < 2; ++tn)
                acc[tm][tn] = wmma16(a[tm], b[tn], acc[tm][tn]);
#if HAS_TDM
        tdm_wait0();          // ensure prefetch landed before buffers swap
        __syncthreads();
#endif
    }

    // epilogue: bias, f16 convert, scatter to attention layouts
#pragma unroll
    for (int tm = 0; tm < 2; ++tm) {
#pragma unroll
        for (int tn = 0; tn < 2; ++tn) {
            const int nl = nl0 + cn + tn * 16 + lane16;
            const float bv_ = bias[nl];
            const int h_ = nl >> 6, d_ = nl & 63;
            const int mbase = m0 + rm + tm * 16 + 8 * half;
            const int bb = mbase >> 11, s0 = mbase & 2047;
            const int bh = bb * H_ + h_;
            if (mat < 2) {
                _Float16* dst = (mat == 0) ? Qh : Kh;
#pragma unroll
                for (int r = 0; r < 8; ++r)
                    dst[((size_t)bh * S_ + (s0 + r)) * DK_ + d_] =
                        (_Float16)(acc[tm][tn][r] + bv_);
            } else {
                v8h pk;
#pragma unroll
                for (int r = 0; r < 8; ++r)
                    pk[r] = (_Float16)(acc[tm][tn][r] + bv_);
                *(v8h*)(VTh + ((size_t)bh * DK_ + d_) * S_ + s0) = pk;
            }
        }
    }
}

// ---------------------------------------------------------------------------
// Kernel 2: flash attention per (b,h).  K/V^T tiles staged once per block by
// the TDM (shared by all 8 waves), double buffered.  Each wave: 16 queries.
// grid = (B*H, S/128), block = 256.
// ---------------------------------------------------------------------------
__global__ __launch_bounds__(256) void flash_attn_wmma(
    const _Float16* __restrict__ Qh, const _Float16* __restrict__ Kh,
    const _Float16* __restrict__ VTh, _Float16* __restrict__ CTXh)
{
    const int tid    = threadIdx.x;
    const int lane   = tid & 31;
    const int wid    = tid >> 5;
    const int lane16 = lane & 15;
    const int half   = lane >> 4;

    const int bh = blockIdx.x;
    const int q0 = blockIdx.y * 128 + wid * 16;
    const size_t kbase = (size_t)bh * S_ * DK_;   // K rows (key, dk)
    const size_t vbase = (size_t)bh * DK_ * S_;   // V^T rows (dk, s)

    __shared__ __align__(16) _Float16 Ks[2][32 * KPAD];    // 32 keys x 64 dk
    __shared__ __align__(16) _Float16 Vs[2][64 * LDSPAD];  // 64 dk x 32 keys
    __shared__ __align__(16) _Float16 Plds[8 * 16 * LDSPAD];
    _Float16* P = Plds + wid * 16 * LDSPAD;

    const _Float16* qp = Qh + kbase + (size_t)(q0 + lane16) * DK_;
    const v16h qa0 = mk16(*(const v8h*)(qp + half * 8),
                          *(const v8h*)(qp + 16 + half * 8));
    const v16h qa1 = mk16(*(const v8h*)(qp + 32 + half * 8),
                          *(const v8h*)(qp + 48 + half * 8));

    v8f ctx[4];
#pragma unroll
    for (int t = 0; t < 4; ++t) ctx[t] = vzero8f();
    float mrun[8], lrun[8];
#pragma unroll
    for (int r = 0; r < 8; ++r) { mrun[r] = -1e30f; lrun[r] = 0.0f; }

#if HAS_TDM
    if (wid == 0) {
        tdm_load_2d(ldsoff(Ks[0]), Kh + kbase, 64, 32, 64, 4, 3);
        tdm_load_2d(ldsoff(Vs[0]), VTh + vbase, 32, 64, S_, 3, 3);
    }
    tdm_wait0();
    __syncthreads();
#endif
    for (int i = 0; i < S_ / 32; ++i) {
        const int kb = i * 32;
#if HAS_TDM
        const int cur = i & 1;
        if (i + 1 < S_ / 32 && wid == 0) {
            tdm_load_2d(ldsoff(Ks[cur ^ 1]), Kh + kbase + (size_t)(kb + 32) * DK_,
                        64, 32, 64, 4, 3);
            tdm_load_2d(ldsoff(Vs[cur ^ 1]), VTh + vbase + kb + 32,
                        32, 64, S_, 3, 3);
        }
        const _Float16* Ksb = Ks[cur];
        const _Float16* Vsb = Vs[cur];
#else
        __syncthreads();
        {   // K tile: 32 rows x 64 f16, 8 thr/row
            const int row = tid >> 3, kj = (tid & 7) * 8;
            *(v8h*)(Ks[0] + row * KPAD + kj) =
                *(const v8h*)(Kh + kbase + (size_t)(kb + row) * DK_ + kj);
        }
        {   // V^T tile: 64 rows x 32 f16, 4 thr/row
            const int row = tid >> 2, kj = (tid & 3) * 8;
            *(v8h*)(Vs[0] + row * LDSPAD + kj) =
                *(const v8h*)(VTh + vbase + (size_t)row * S_ + kb + kj);
        }
        __syncthreads();
        const _Float16* Ksb = Ks[0];
        const _Float16* Vsb = Vs[0];
#endif
        // ---- scores: Q(16x64) . K^T(64x32) ----
        v8f sc[2];
#pragma unroll
        for (int kt = 0; kt < 2; ++kt) {
            const _Float16* kp = Ksb + (kt * 16 + lane16) * KPAD;
            v16h b0 = mk16(*(const v8h*)(kp + half * 8),
                           *(const v8h*)(kp + 16 + half * 8));
            v16h b1 = mk16(*(const v8h*)(kp + 32 + half * 8),
                           *(const v8h*)(kp + 48 + half * 8));
            v8f c = vzero8f();
            c = wmma16(qa0, b0, c);
            c = wmma16(qa1, b1, c);
            sc[kt] = c;
        }
        // ---- online softmax over 32 keys ----
#pragma unroll
        for (int r = 0; r < 8; ++r) {
            float s0 = sc[0][r] * 0.125f;
            float s1 = sc[1][r] * 0.125f;
            float mx = fmaxf(s0, s1);
#pragma unroll
            for (int off = 1; off < 16; off <<= 1)
                mx = fmaxf(mx, __shfl_xor(mx, off, 16));
            const float mnew = fmaxf(mrun[r], mx);
            const float p0 = __expf(s0 - mnew);
            const float p1 = __expf(s1 - mnew);
            float rs = p0 + p1;
#pragma unroll
            for (int off = 1; off < 16; off <<= 1)
                rs += __shfl_xor(rs, off, 16);
            const float alpha = __expf(mrun[r] - mnew);
            lrun[r] = lrun[r] * alpha + rs;
            mrun[r] = mnew;
#pragma unroll
            for (int t = 0; t < 4; ++t) ctx[t][r] *= alpha;
            const int prow = r + 8 * half;
            P[prow * LDSPAD + lane16]      = (_Float16)p0;
            P[prow * LDSPAD + 16 + lane16] = (_Float16)p1;
        }
        // re-read P in A-fragment layout (wave-local; DS pipe is in-order)
        const _Float16* pr = P + lane16 * LDSPAD;
        const v16h pa = mk16(*(const v8h*)(pr + half * 8),
                             *(const v8h*)(pr + 16 + half * 8));
        // ---- ctx += P(16x32) . V(32x64) ----
#pragma unroll
        for (int tn = 0; tn < 4; ++tn) {
            const _Float16* vp = Vsb + (tn * 16 + lane16) * LDSPAD + half * 8;
            v16h vb = mk16(*(const v8h*)vp, *(const v8h*)(vp + 16));
            ctx[tn] = wmma16(pa, vb, ctx[tn]);
        }
#if HAS_TDM
        tdm_wait0();
        __syncthreads();
#endif
    }

    const int bb = bh >> 3, h_ = bh & 7;
#pragma unroll
    for (int tn = 0; tn < 4; ++tn) {
        const int d_ = tn * 16 + lane16;
#pragma unroll
        for (int r = 0; r < 8; ++r) {
            const float v = ctx[tn][r] / lrun[r];
            const int qq = q0 + r + 8 * half;
            CTXh[((size_t)(bb * S_ + qq)) * D_ + h_ * DK_ + d_] = (_Float16)v;
        }
    }
}

// ---------------------------------------------------------------------------
// Kernel 3: out = ctx @ Wo^T + bo; y = x + out; LayerNorm(y)*gamma + beta.
// grid = M/32, block = 256 (8 waves x 64 cols = full 512-wide row for LN).
// Tiles staged by TDM (single buffer: Wo is L2-resident).
// ---------------------------------------------------------------------------
__global__ __launch_bounds__(256) void proj_ln_wmma(
    const _Float16* __restrict__ CTXh, const _Float16* __restrict__ Woh,
    const float* __restrict__ x, const float* __restrict__ bo,
    const float* __restrict__ gamma, const float* __restrict__ beta,
    float* __restrict__ out)
{
    const int tid    = threadIdx.x;
    const int lane   = tid & 31;
    const int wid    = tid >> 5;
    const int lane16 = lane & 15;
    const int half   = lane >> 4;

    const int m0 = blockIdx.x * 32;
    const int cn = wid * 64;

    __shared__ __align__(16) _Float16 As[32 * LDSPAD];
    __shared__ __align__(16) _Float16 Bs[512 * LDSPAD];
    __shared__ float rsum[32], rsq[32];
    if (tid < 32) { rsum[tid] = 0.0f; rsq[tid] = 0.0f; }

    v8f acc[2][4];
#pragma unroll
    for (int i = 0; i < 2; ++i)
#pragma unroll
        for (int j = 0; j < 4; ++j) acc[i][j] = vzero8f();

    for (int kk = 0; kk < 512; kk += 32) {
        __syncthreads();
#if HAS_TDM
        if (wid == 0) {
            tdm_load_2d(ldsoff(As), CTXh + (size_t)m0 * 512 + kk, 32,  32, 512, 3, 3);
            tdm_load_2d(ldsoff(Bs), Woh + kk,                     32, 512, 512, 3, 3);
        }
        tdm_wait0();
#else
        {   // ctx tile: 32 rows x 32 f16
            const int row = tid >> 3, kj = (tid & 7) * 4;
            *(v4h*)(As + row * LDSPAD + kj) =
                *(const v4h*)(CTXh + (size_t)(m0 + row) * 512 + kk + kj);
        }
#pragma unroll
        for (int rr = 0; rr < 2; ++rr) {   // Wo tile: 512 rows x 32 f16
            const int row = tid * 2 + rr;
            const _Float16* src = Woh + (size_t)row * 512 + kk;
#pragma unroll
            for (int j = 0; j < 32; j += 8)
                *(v8h*)(Bs + row * LDSPAD + j) = *(const v8h*)(src + j);
        }
#endif
        __syncthreads();

        v16h a[2], b[4];
#pragma unroll
        for (int tm = 0; tm < 2; ++tm) {
            const _Float16* p = As + (tm * 16 + lane16) * LDSPAD;
            a[tm] = mk16(*(const v8h*)(p + half * 8),
                         *(const v8h*)(p + 16 + half * 8));
        }
#pragma unroll
        for (int tn = 0; tn < 4; ++tn) {
            const _Float16* p = Bs + (cn + tn * 16 + lane16) * LDSPAD + half * 8;
            b[tn] = mk16(*(const v8h*)p, *(const v8h*)(p + 16));
        }
#pragma unroll
        for (int tm = 0; tm < 2; ++tm)
#pragma unroll
            for (int tn = 0; tn < 4; ++tn)
                acc[tm][tn] = wmma16(a[tm], b[tn], acc[tm][tn]);
    }

    // ---- epilogue: y = acc + bo + x, then LayerNorm across D=512 ----
    float bon[4], gn[4], bn[4];
#pragma unroll
    for (int tn = 0; tn < 4; ++tn) {
        const int n = cn + tn * 16 + lane16;
        bon[tn] = bo[n]; gn[tn] = gamma[n]; bn[tn] = beta[n];
    }
#pragma unroll
    for (int tm = 0; tm < 2; ++tm)
#pragma unroll
        for (int r = 0; r < 8; ++r) {
            const int m = m0 + tm * 16 + r + 8 * half;
#pragma unroll
            for (int tn = 0; tn < 4; ++tn) {
                const int n = cn + tn * 16 + lane16;
                acc[tm][tn][r] += bon[tn] + x[(size_t)m * 512 + n];
            }
        }

#pragma unroll
    for (int tm = 0; tm < 2; ++tm)
#pragma unroll
        for (int r = 0; r < 8; ++r) {
            float s1 = 0.0f, s2 = 0.0f;
#pragma unroll
            for (int tn = 0; tn < 4; ++tn) {
                const float v = acc[tm][tn][r];
                s1 += v; s2 += v * v;
            }
#pragma unroll
            for (int off = 1; off < 16; off <<= 1) {
                s1 += __shfl_xor(s1, off, 16);
                s2 += __shfl_xor(s2, off, 16);
            }
            if (lane16 == 0) {
                const int row = tm * 16 + r + 8 * half;
                atomicAdd(&rsum[row], s1);   // ds_add_f32
                atomicAdd(&rsq[row],  s2);
            }
        }
    __syncthreads();

#pragma unroll
    for (int tm = 0; tm < 2; ++tm)
#pragma unroll
        for (int r = 0; r < 8; ++r) {
            const int row = tm * 16 + r + 8 * half;
            const float mu   = rsum[row] * (1.0f / 512.0f);
            const float var  = rsq[row] * (1.0f / 512.0f) - mu * mu;
            const float rstd = rsqrtf(var + 1e-5f);
            const int m = m0 + row;
#pragma unroll
            for (int tn = 0; tn < 4; ++tn) {
                const int n = cn + tn * 16 + lane16;
                out[(size_t)m * 512 + n] =
                    (acc[tm][tn][r] - mu) * rstd * gn[tn] + bn[tn];
            }
        }
}

// ---------------------------------------------------------------------------
extern "C" void kernel_launch(void* const* d_in, const int* in_sizes, int n_in,
                              void* d_out, int out_size, void* d_ws, size_t ws_size,
                              hipStream_t stream) {
    const float* x     = (const float*)d_in[0];
    const float* Wq    = (const float*)d_in[1];
    const float* bq    = (const float*)d_in[2];
    const float* Wk    = (const float*)d_in[3];
    const float* bk    = (const float*)d_in[4];
    const float* Wv    = (const float*)d_in[5];
    const float* bv    = (const float*)d_in[6];
    const float* Wo    = (const float*)d_in[7];
    const float* bo    = (const float*)d_in[8];
    const float* gamma = (const float*)d_in[9];
    const float* beta  = (const float*)d_in[10];
    float* out = (float*)d_out;

    // workspace (all f16): Q,K (B,H,S,dk); V^T (B,H,dk,S); ctx (B,S,D);
    // Xh (M,D); Wqh/Wkh/Wvh/Woh (D,D)
    _Float16* Qh   = (_Float16*)d_ws;
    _Float16* Kh   = Qh   + (size_t)M_ * D_;
    _Float16* VTh  = Kh   + (size_t)M_ * D_;
    _Float16* CTXh = VTh  + (size_t)M_ * D_;
    _Float16* Xh   = CTXh + (size_t)M_ * D_;
    _Float16* Wqh  = Xh   + (size_t)M_ * D_;
    _Float16* Wkh  = Wqh  + (size_t)D_ * D_;
    _Float16* Wvh  = Wkh  + (size_t)D_ * D_;
    _Float16* Woh  = Wvh  + (size_t)D_ * D_;

    cvt_x_f16<<<dim3((M_ * D_) / 1024), 256, 0, stream>>>(x, Xh);
    cvt_w_f16<<<dim3((D_ * D_) / 1024, 4), 256, 0, stream>>>(
        Wq, Wk, Wv, Wo, Wqh, Wkh, Wvh, Woh);

    qkv_proj_wmma<<<dim3(M_ / 64, (3 * D_) / 128), 256, 0, stream>>>(
        Xh, Wqh, Wkh, Wvh, bq, bk, bv, Qh, Kh, VTh);

    flash_attn_wmma<<<dim3(B_ * H_, S_ / 128), 256, 0, stream>>>(
        Qh, Kh, VTh, CTXh);

    proj_ln_wmma<<<dim3(M_ / 32), 256, 0, stream>>>(
        CTXh, Woh, x, bo, gamma, beta, out);
}
